// SAlibi4Attention_86552180949669
// MI455X (gfx1250) — matrix operand order repaired and checked by reading, not compiled, not run on hardware
//
#include <hip/hip_runtime.h>
#include <math.h>

// Problem constants (from the reference): B=2, L=2048, D=1024, H=16, DH=64.
#define BB  2
#define LLQ 2048
#define DD  1024
#define HH  16
#define DHH 64
#define BL  4096   // B*L

typedef __attribute__((ext_vector_type(16))) __bf16  v16bf;
typedef __attribute__((ext_vector_type(2)))  __bf16  v2bf;
typedef __attribute__((ext_vector_type(8)))  float   v8f;
typedef __attribute__((ext_vector_type(4)))  float   v4f;
typedef __attribute__((ext_vector_type(4)))  unsigned int v4u;

union Frag16 { v16bf v; v4u q[2]; v2bf pk[8]; unsigned short us[16]; };

// f32 pair -> packed bf16 (RTNE). Native packed convert (verified: v_cvt_pk_bf16_f32).
__device__ __forceinline__ v2bf pk2(float x, float y) {
#if __has_builtin(__builtin_amdgcn_cvt_pk_bf16_f32)
  return __builtin_amdgcn_cvt_pk_bf16_f32(x, y);
#else
  v2bf r; r[0] = (__bf16)x; r[1] = (__bf16)y; return r;
#endif
}

__device__ __forceinline__ unsigned short f2bf(float x) {
  union { __bf16 h; unsigned short u; } c;
  c.h = (__bf16)x;
  return c.u;
}

__device__ __forceinline__ v8f wmma_bf16(v16bf a, v16bf b, v8f c) {
  // D = A(16x32 bf16) * B(32x16 bf16) + C(16x16 f32)
  return __builtin_amdgcn_wmma_f32_16x16x32_bf16(
      /*neg_a=*/false, a, /*neg_b=*/false, b,
      /*c_mod=*/(short)0, c, /*reuse_a=*/false, /*reuse_b=*/false);
}

// ---- A fragment (16x32, MxK): lane 0-15 -> row M=lane, K = {0..7, 16..23};
//      lane 16-31 -> row M=lane-16, K = {8..15, 24..31}.
__device__ __forceinline__ v16bf loadA_bf16(const unsigned short* base, int ld, int row0, int kbase) {
  const int lane = threadIdx.x & 31;
  const int lh = lane >> 4, r = lane & 15;
  const unsigned short* p = base + (size_t)(row0 + r) * ld + kbase + lh * 8;
  Frag16 f;
  f.q[0] = *(const v4u*)(p);        // K {0..7} (or {8..15})
  f.q[1] = *(const v4u*)(p + 16);   // K {16..23} (or {24..31})
  return f.v;
}

// ---- B fragment (32x16, KxN): lane&15 = column N; lanes 0-15 hold K={0..7,16..23},
//      lanes 16-31 hold K={8..15,24..31} (matches A). Column n = row (col0+n) of source.
__device__ __forceinline__ v16bf loadB_bf16(const unsigned short* base, int ld, int col0, int kbase) {
  const int lane = threadIdx.x & 31;
  const int lh = lane >> 4, n = lane & 15;
  const unsigned short* p = base + (size_t)(col0 + n) * ld + kbase + lh * 8;
  Frag16 f;
  f.q[0] = *(const v4u*)(p);
  f.q[1] = *(const v4u*)(p + 16);
  return f.v;
}

// ===================== 0) one-shot f32 -> bf16 conversion =====================
// Converts X / W operands once so the GEMM hot loops carry zero VALU converts
// and half the bytes. Bandwidth-bound: ~80MB @ 23.3 TB/s ~= 3.5us.
__global__ __launch_bounds__(256) void cvt_bf16_kernel(const float* __restrict__ src,
                                                       unsigned short* __restrict__ dst,
                                                       int n8) {
  const int i = blockIdx.x * 256 + threadIdx.x;
  if (i >= n8) return;
  const v4f a0 = *(const v4f*)(src + (size_t)i * 8);
  const v4f a1 = *(const v4f*)(src + (size_t)i * 8 + 4);
  union { v4u q; v2bf pk[4]; } o;
  o.pk[0] = pk2(a0.x, a0.y); o.pk[1] = pk2(a0.z, a0.w);
  o.pk[2] = pk2(a1.x, a1.y); o.pk[3] = pk2(a1.z, a1.w);
  *(v4u*)(dst + (size_t)i * 8) = o.q;
}

// ===================== 1) fused QKV projection =====================
// out = X @ W^T + b; each wave computes a 32x64 tile (2 M-tiles x one head in N).
// z=0: Q -> [B,H,L,DH]   z=1: K -> [B,H,L,DH]   z=2: V -> [B,H,DH,L] (transposed)
__global__ __launch_bounds__(32) void qkv_proj_kernel(
    const unsigned short* __restrict__ Xq, const unsigned short* __restrict__ Xk,
    const unsigned short* __restrict__ Xv,
    const unsigned short* __restrict__ Wqb, const float* __restrict__ bq,
    const unsigned short* __restrict__ Wkb, const float* __restrict__ bk,
    const unsigned short* __restrict__ Wvb, const float* __restrict__ bv,
    unsigned short* __restrict__ Qh, unsigned short* __restrict__ Kh,
    unsigned short* __restrict__ Vt) {
  const int z = blockIdx.z;
  const unsigned short* X = (z == 0) ? Xq : (z == 1) ? Xk : Xv;
  const unsigned short* W = (z == 0) ? Wqb : (z == 1) ? Wkb : Wvb;
  const float* bias = (z == 0) ? bq : (z == 1) ? bk : bv;
  const int row0 = blockIdx.x * 32;
  const int h    = blockIdx.y;        // 64-wide N block == one head
  const int col0 = h * DHH;
  const int lane = threadIdx.x & 31;
  const int ln = lane & 15, lh = lane >> 4;

  v8f acc[2][4];
#pragma unroll
  for (int m = 0; m < 2; ++m)
#pragma unroll
    for (int t = 0; t < 4; ++t)
#pragma unroll
      for (int i = 0; i < 8; ++i) acc[m][t][i] = 0.0f;

  for (int kb = 0; kb < DD; kb += 32) {
    __builtin_prefetch(X + (size_t)(row0 + ln) * DD + kb + 32, 0, 3);  // WGP-scope
    __builtin_prefetch(W + (size_t)(col0 + ln) * DD + kb + 32, 0, 3);
    const v16bf a0 = loadA_bf16(X, DD, row0, kb);
    const v16bf a1 = loadA_bf16(X, DD, row0 + 16, kb);
#pragma unroll
    for (int t = 0; t < 4; ++t) {
      const v16bf bfr = loadB_bf16(W, DD, col0 + t * 16, kb);
      acc[0][t] = wmma_bf16(a0, bfr, acc[0][t]);
      acc[1][t] = wmma_bf16(a1, bfr, acc[1][t]);
    }
  }

  // C layout: reg r -> row (r, or r+8 for upper half-wave); lane&15 -> column
#pragma unroll
  for (int mt = 0; mt < 2; ++mt) {
    const int mbase = row0 + mt * 16 + (lh << 3);
#pragma unroll
    for (int t = 0; t < 4; ++t) {
      const int d = t * 16 + ln;                  // position within head
      const float bcol = bias[col0 + d];
#pragma unroll
      for (int r = 0; r < 8; ++r) {
        const int m = mbase + r;                  // global row in [0, B*L)
        const int bb = m >> 11, l = m & (LLQ - 1);
        const unsigned short val = f2bf(acc[mt][t][r] + bcol);
        if (z == 2) {
          Vt[((size_t)(bb * HH + h) * DHH + d) * LLQ + l] = val;
        } else {
          const size_t idx = ((size_t)(bb * HH + h) * LLQ + l) * DHH + d;
          if (z == 0) Qh[idx] = val; else Kh[idx] = val;
        }
      }
    }
  }
}

// ===================== 2) flash attention with ALiBi-softplus bias =====================
// one wave per (b, h, 32-query block): 2 query tiles share every K/V fragment,
// so each 32-key chunk runs 16 WMMAs on 16 b128 fragment loads.
__global__ __launch_bounds__(32) void attn_kernel(
    const unsigned short* __restrict__ Qh, const unsigned short* __restrict__ Kh,
    const unsigned short* __restrict__ Vt, unsigned short* __restrict__ Ctx) {
  const int b = blockIdx.z, h = blockIdx.y, qt = blockIdx.x;
  const int lane = threadIdx.x & 31;
  const int lh = lane >> 4, ln = lane & 15;
  const int q0 = qt * 32;

  const unsigned short* Qb = Qh + (size_t)(b * HH + h) * LLQ * DHH;  // [L][64]
  const unsigned short* Kb = Kh + (size_t)(b * HH + h) * LLQ * DHH;  // [L][64]
  const unsigned short* Vb = Vt + (size_t)(b * HH + h) * DHH * LLQ;  // [64][L]

  // m_h = exp(-5000^(h/15 * 0.8)); underflows to 0 for large h, same as reference
  const float mh = expf(-powf(5000.0f, ((float)h / 15.0f) * 0.8f));

  __shared__ unsigned short Plds[2][16][32];   // C-layout -> A-layout staging

  v16bf qa[2][2];
#pragma unroll
  for (int mt = 0; mt < 2; ++mt) {
    qa[mt][0] = loadA_bf16(Qb, DHH, q0 + mt * 16, 0);
    qa[mt][1] = loadA_bf16(Qb, DHH, q0 + mt * 16, 32);
  }

  v8f o[2][4];
  float mrun[2][8], srun[2][8];
#pragma unroll
  for (int mt = 0; mt < 2; ++mt)
#pragma unroll
    for (int i = 0; i < 8; ++i) {
      o[mt][0][i] = o[mt][1][i] = o[mt][2][i] = o[mt][3][i] = 0.0f;
      mrun[mt][i] = -1e30f; srun[mt][i] = 0.0f;
    }

  for (int jc = 0; jc < LLQ; jc += 32) {
    __builtin_prefetch(Kb + (size_t)(jc + 32 + ln) * DHH, 0, 3);
    __builtin_prefetch(Vb + (size_t)ln * LLQ + jc + 32, 0, 3);

    // K fragments for this 32-key chunk, shared by both query tiles
    const v16bf kb00 = loadB_bf16(Kb, DHH, jc, 0);
    const v16bf kb01 = loadB_bf16(Kb, DHH, jc, 32);
    const v16bf kb10 = loadB_bf16(Kb, DHH, jc + 16, 0);
    const v16bf kb11 = loadB_bf16(Kb, DHH, jc + 16, 32);

    const float j0 = (float)(jc + ln), j1 = j0 + 16.0f;

#pragma unroll
    for (int mt = 0; mt < 2; ++mt) {
      // scores: two 16x16 tiles, contraction over DH=64 (2 K-steps)
      v8f s0, s1;
#pragma unroll
      for (int i = 0; i < 8; ++i) { s0[i] = 0.0f; s1[i] = 0.0f; }
      s0 = wmma_bf16(qa[mt][0], kb00, s0);
      s0 = wmma_bf16(qa[mt][1], kb01, s0);
      s1 = wmma_bf16(qa[mt][0], kb10, s1);
      s1 = wmma_bf16(qa[mt][1], kb11, s1);

      // scale 1/sqrt(64) + bias m_h*softplus(1000+dist).
      // 1000+dist >= 1000  =>  softplus(x) == x bit-exactly in fp32.
      const float ibase = (float)(q0 + mt * 16 + lh * 8);
#pragma unroll
      for (int r = 0; r < 8; ++r) {
        const float fi = ibase + (float)r;
        s0[r] = s0[r] * 0.125f + mh * (1000.0f + fabsf(fi - j0));
        s1[r] = s1[r] * 0.125f + mh * (1000.0f + fabsf(fi - j1));
      }

      // online softmax: per-row (reg r) reductions across the 16-lane half
#pragma unroll
      for (int r = 0; r < 8; ++r) {
        float cm = fmaxf(s0[r], s1[r]);
#pragma unroll
        for (int off = 1; off < 16; off <<= 1)
          cm = fmaxf(cm, __shfl_xor(cm, off, 32));
        const float mnew = fmaxf(mrun[mt][r], cm);
        const float scale = __expf(mrun[mt][r] - mnew);
        const float p0 = __expf(s0[r] - mnew);
        const float p1 = __expf(s1[r] - mnew);
        float rs = p0 + p1;
#pragma unroll
        for (int off = 1; off < 16; off <<= 1)
          rs += __shfl_xor(rs, off, 32);
        srun[mt][r] = srun[mt][r] * scale + rs;
        mrun[mt][r] = mnew;
        o[mt][0][r] *= scale; o[mt][1][r] *= scale;
        o[mt][2][r] *= scale; o[mt][3][r] *= scale;
        Plds[mt][r + lh * 8][ln]      = f2bf(p0);
        Plds[mt][r + lh * 8][ln + 16] = f2bf(p1);
      }
    }
    __syncthreads();  // single-wave WG: cheap; orders LDS store -> cross-lane load

    const v16bf pa0 = loadA_bf16(&Plds[0][0][0], 32, 0, 0);  // P tiles as A-frags
    const v16bf pa1 = loadA_bf16(&Plds[1][0][0], 32, 0, 0);
#pragma unroll
    for (int t = 0; t < 4; ++t) {
      const v16bf vb = loadB_bf16(Vb, LLQ, t * 16, jc);      // shared V fragment
      o[0][t] = wmma_bf16(pa0, vb, o[0][t]);
      o[1][t] = wmma_bf16(pa1, vb, o[1][t]);
    }
    __syncthreads();  // protect Plds WAR for next chunk
  }

  // normalize and emit context in [B, L, D] (bf16) for the output projection
#pragma unroll
  for (int mt = 0; mt < 2; ++mt)
#pragma unroll
    for (int r = 0; r < 8; ++r) {
      const int row = q0 + mt * 16 + r + lh * 8;
      const float inv = 1.0f / srun[mt][r];
      const size_t base = ((size_t)(b * LLQ + row)) * DD + h * DHH;
      Ctx[base + 0 * 16 + ln] = f2bf(o[mt][0][r] * inv);
      Ctx[base + 1 * 16 + ln] = f2bf(o[mt][1][r] * inv);
      Ctx[base + 2 * 16 + ln] = f2bf(o[mt][2][r] * inv);
      Ctx[base + 3 * 16 + ln] = f2bf(o[mt][3][r] * inv);
    }
}

// ===================== 3) output projection =====================
// out = ctx @ Wo^T + bo; each wave computes a 32x64 tile, f32 output.
__global__ __launch_bounds__(32) void out_proj_kernel(
    const unsigned short* __restrict__ Ctx, const unsigned short* __restrict__ Wob,
    const float* __restrict__ bo, float* __restrict__ out) {
  const int row0 = blockIdx.x * 32, col0 = blockIdx.y * 64;
  const int lane = threadIdx.x & 31;
  const int ln = lane & 15, lh = lane >> 4;

  v8f acc[2][4];
#pragma unroll
  for (int m = 0; m < 2; ++m)
#pragma unroll
    for (int t = 0; t < 4; ++t)
#pragma unroll
      for (int i = 0; i < 8; ++i) acc[m][t][i] = 0.0f;

  for (int kb = 0; kb < DD; kb += 32) {
    __builtin_prefetch(Ctx + (size_t)(row0 + ln) * DD + kb + 32, 0, 3);
    __builtin_prefetch(Wob + (size_t)(col0 + ln) * DD + kb + 32, 0, 3);
    const v16bf a0 = loadA_bf16(Ctx, DD, row0, kb);
    const v16bf a1 = loadA_bf16(Ctx, DD, row0 + 16, kb);
#pragma unroll
    for (int t = 0; t < 4; ++t) {
      const v16bf bfr = loadB_bf16(Wob, DD, col0 + t * 16, kb);
      acc[0][t] = wmma_bf16(a0, bfr, acc[0][t]);
      acc[1][t] = wmma_bf16(a1, bfr, acc[1][t]);
    }
  }

#pragma unroll
  for (int mt = 0; mt < 2; ++mt) {
    const int mbase = row0 + mt * 16 + (lh << 3);
#pragma unroll
    for (int t = 0; t < 4; ++t) {
      const int n = col0 + t * 16 + ln;
      const float bcol = bo[n];
#pragma unroll
      for (int r = 0; r < 8; ++r)
        out[(size_t)(mbase + r) * DD + n] = acc[mt][t][r] + bcol;
    }
  }
}

extern "C" void kernel_launch(void* const* d_in, const int* in_sizes, int n_in,
                              void* d_out, int out_size, void* d_ws, size_t ws_size,
                              hipStream_t stream) {
  const float* q  = (const float*)d_in[0];
  const float* k  = (const float*)d_in[1];
  const float* v  = (const float*)d_in[2];
  const float* Wq = (const float*)d_in[3];
  const float* bq = (const float*)d_in[4];
  const float* Wk = (const float*)d_in[5];
  const float* bk = (const float*)d_in[6];
  const float* Wv = (const float*)d_in[7];
  const float* bv = (const float*)d_in[8];
  const float* Wo = (const float*)d_in[9];
  const float* bo = (const float*)d_in[10];
  float* out = (float*)d_out;

  // workspace layout (all bf16/ushort):
  //   Qh, Kh        [B,H,L,DH]    4,194,304 each
  //   Vt            [B,H,DH,L]    4,194,304
  //   Ctx           [B,L,D]       4,194,304
  //   Xq, Xk, Xv    [B*L, D]      4,194,304 each
  //   Wqb,Wkb,Wvb,Wob [D, D]      1,048,576 each        total 64 MB
  const size_t perBig = (size_t)BB * HH * LLQ * DHH;   // 4,194,304
  const size_t perW   = (size_t)DD * DD;               // 1,048,576
  unsigned short* Qh  = (unsigned short*)d_ws;
  unsigned short* Kh  = Qh + perBig;
  unsigned short* Vt  = Kh + perBig;
  unsigned short* Ctx = Vt + perBig;
  unsigned short* Xq  = Ctx + perBig;
  unsigned short* Xk  = Xq + perBig;
  unsigned short* Xv  = Xk + perBig;
  unsigned short* Wqb = Xv + perBig;
  unsigned short* Wkb = Wqb + perW;
  unsigned short* Wvb = Wkb + perW;
  unsigned short* Wob = Wvb + perW;

  // one-shot conversions (bandwidth-bound)
  const int nBig8 = (int)(perBig / 8), nW8 = (int)(perW / 8);
  cvt_bf16_kernel<<<(nBig8 + 255) / 256, 256, 0, stream>>>(q, Xq, nBig8);
  cvt_bf16_kernel<<<(nBig8 + 255) / 256, 256, 0, stream>>>(k, Xk, nBig8);
  cvt_bf16_kernel<<<(nBig8 + 255) / 256, 256, 0, stream>>>(v, Xv, nBig8);
  cvt_bf16_kernel<<<(nW8 + 255) / 256, 256, 0, stream>>>(Wq, Wqb, nW8);
  cvt_bf16_kernel<<<(nW8 + 255) / 256, 256, 0, stream>>>(Wk, Wkb, nW8);
  cvt_bf16_kernel<<<(nW8 + 255) / 256, 256, 0, stream>>>(Wv, Wvb, nW8);
  cvt_bf16_kernel<<<(nW8 + 255) / 256, 256, 0, stream>>>(Wo, Wob, nW8);

  qkv_proj_kernel<<<dim3(BL / 32, HH, 3), 32, 0, stream>>>(
      Xq, Xk, Xv, Wqb, bq, Wkb, bk, Wvb, bv, Qh, Kh, Vt);
  attn_kernel<<<dim3(LLQ / 32, HH, BB), 32, 0, stream>>>(Qh, Kh, Vt, Ctx);
  out_proj_kernel<<<dim3(BL / 32, DD / 64), 32, 0, stream>>>(Ctx, Wob, bo, out);
}
